// DMM_36301063585939
// MI455X (gfx1250) — matrix operand order repaired and checked by reading, not compiled, not run on hardware
//
#include <hip/hip_runtime.h>

#define TT 512
#define BB 256
#define DD 88
#define ZZ 100

typedef __attribute__((ext_vector_type(16))) _Float16 v16h;
typedef __attribute__((ext_vector_type(8)))  _Float16 v8h;
typedef __attribute__((ext_vector_type(8)))  float    v8f;
typedef __attribute__((ext_vector_type(4)))  unsigned int v4u;
typedef __attribute__((ext_vector_type(8)))  int v8i;
typedef __attribute__((ext_vector_type(4)))  int v4i;

// ---------------- WMMA helpers (v_wmma_f32_16x16x32_f16) ----------------

// A-matrix fragment (16x32 f16, row-major source, stride lda):
// lane L: row = m0 + (L&15), half = L>>4; elements i = K[16h + i]
// (contiguous, matching the B fragment)
__device__ __forceinline__ v16h ldfragA(const _Float16* p) {
  union { v16h v; v8h h[2]; } u;
  u.h[0] = *(const v8h*)(p);
  u.h[1] = *(const v8h*)(p + 8);
  return u.v;
}

// B-matrix fragment (32x16 f16). B[k][n] = W[n][k] with W row-major (NxK):
// lane L: col = n0 + (L&15), half = L>>4; elements i = K[16h + i] -> one
// contiguous 32-byte load from the weight row.
__device__ __forceinline__ v16h ldfragB(const _Float16* p) {
  return *(const v16h*)p;
}

__device__ __forceinline__ v8f wmma_f16(v16h a, v16h b, v8f c) {
  return __builtin_amdgcn_wmma_f32_16x16x32_f16(false, a, false, b, (short)0, c,
                                                false, false);
}

// Accumulate A(16xK) * W^T(Kx16) over ktiles K-tiles of 32.
// arow = m0 + (lane&15)  (A row for this lane)
// wrow = n0 + (lane&15)  (weight row == output column for this lane)
__device__ __forceinline__ v8f gemm_acc(const _Float16* __restrict__ A, int lda,
                                        const _Float16* __restrict__ W, int ldw,
                                        int arow, int wrow, int ktiles,
                                        v8f acc, int hf) {
  const _Float16* pa = A + (size_t)arow * lda + hf * 16;
  const _Float16* pw = W + (size_t)wrow * ldw + hf * 16;
  for (int k = 0; k < ktiles; ++k) {
    acc = wmma_f16(ldfragA(pa), ldfragB(pw), acc);
    pa += 32;
    pw += 32;
  }
  return acc;
}

// sum across the 16 lanes of one half-wave (columns of a C tile row)
__device__ __forceinline__ float rowsum16(float x) {
  x += __shfl_xor(x, 8, 16);
  x += __shfl_xor(x, 4, 16);
  x += __shfl_xor(x, 2, 16);
  x += __shfl_xor(x, 1, 16);
  return x;
}

__device__ __forceinline__ float softplusf(float x) {
  return (x > 20.f) ? x : log1pf(expf(x));
}
__device__ __forceinline__ float sigmoidf(float x) {
  return 1.f / (1.f + expf(-x));
}
#define HALF_LOG_2PI 0.918938533204672742f

// ---------------- setup kernels ----------------

__global__ void k_cvt_pad(const float* __restrict__ src, _Float16* __restrict__ dst,
                          int rows, int cols, int rpad, int cwrite, int dstride) {
  int i = blockIdx.x * 256 + threadIdx.x;
  if (i >= rpad * cwrite) return;
  int r = i / cwrite, c = i % cwrite;
  float v = (r < rows && c < cols) ? src[(size_t)r * cols + c] : 0.f;
  dst[(size_t)r * dstride + c] = (_Float16)v;
}

__global__ void k_pad_bias(const float* __restrict__ src, float* __restrict__ dst,
                           int n, int npad) {
  int i = blockIdx.x * 256 + threadIdx.x;
  if (i >= npad) return;
  dst[i] = (i < n) ? src[i] : 0.f;
}

__global__ void k_bias2(const float* __restrict__ a, const float* __restrict__ b,
                        float* __restrict__ dst, int n, int npad) {
  int i = blockIdx.x * 256 + threadIdx.x;
  if (i >= npad) return;
  dst[i] = (i < n) ? (a[i] + b[i]) : 0.f;
}

// mini_batch (B,T,D) f32 -> X16 (T,B,96) f16, zero padded
__global__ void k_cvt_x(const float* __restrict__ mb, _Float16* __restrict__ X) {
  size_t i = (size_t)blockIdx.x * 256 + threadIdx.x;
  if (i >= (size_t)TT * BB * 96) return;
  int d = (int)(i % 96);
  size_t tm = i / 96;
  int m = (int)(tm % BB);
  int t = (int)(tm / BB);
  float v = (d < DD) ? mb[((size_t)m * TT + t) * DD + d] : 0.f;
  X[i] = (_Float16)v;
}

__global__ void k_bcast_h0(const float* __restrict__ h0, _Float16* __restrict__ H) {
  int i = blockIdx.x * 256 + threadIdx.x;
  if (i >= BB * 608) return;
  int n = i % 608;
  H[i] = (_Float16)((n < 600) ? h0[n] : 0.f);
}

__global__ void k_bcast_z0(const float* __restrict__ z0, _Float16* __restrict__ z16,
                           float* __restrict__ z32) {
  int i = blockIdx.x * 256 + threadIdx.x;
  if (i >= BB * 128) return;
  int n = i % 128;
  float v = (n < ZZ) ? z0[n] : 0.f;
  z16[i] = (_Float16)v;
  z32[i] = v;
}

// ---------------- phase 1: backward RNN step ----------------
// Hout = relu([X_t | Hprev] @ Wc^T + bc), M=256, N=608(600), K=96+608
// All 8 waves of a WG share one N-tile (ni = blockIdx>>1), so the 16 weight
// rows (16x704 f16 = 22.5KB) are staged into LDS once per WG by the Tensor
// Data Mover, then B-fragments are read from LDS (8x less L2 traffic).

struct RnnP {
  const _Float16 *X, *H, *Wc;
  const float* bc;
  _Float16* Hout;
};

__global__ void k_rnn(RnnP p) {
  __shared__ _Float16 lw[16 * 704];  // weight stripe rows n0..n0+15
  int lane = threadIdx.x & 31;
  int tile = blockIdx.x * 8 + (threadIdx.x >> 5);  // 608 tiles exactly
  int li = lane & 15, hf = lane >> 4;
  int m0 = (tile & 15) * 16;
  int n0 = (tile >> 4) * 16;  // uniform across the workgroup
  int col = n0 + li;

  if (threadIdx.x < 32) {  // wave 0 issues one TDM load: 16 rows x 704 cols f16
    unsigned long long ga = (unsigned long long)(size_t)(p.Wc + (size_t)n0 * 704);
    unsigned int lds_off = (unsigned int)(size_t)(void*)lw;  // low 32 = LDS offset
    v4u g0;
    g0[0] = 1u;                                   // count=1, user descriptor
    g0[1] = lds_off;                              // lds_addr
    g0[2] = (unsigned int)(ga & 0xFFFFFFFFu);     // global_addr[31:0]
    g0[3] = (unsigned int)((ga >> 32) & 0x1FFFFFFu) | (2u << 30);  // addr hi|type=2
    v8i g1;
    g1[0] = 0x00010000;             // wg_mask=0, data_size=1 (2 bytes)
    g1[1] = (int)((704u & 0xFFFFu) << 16);  // tensor_dim0 lo16 (=704)
    g1[2] = (int)(16u << 16);       // tensor_dim0 hi=0 | tensor_dim1 lo16 = 16
    g1[3] = (int)(704u << 16);      // tensor_dim1 hi=0 | tile_dim0 = 704
    g1[4] = 16;                     // tile_dim1 = 16, tile_dim2 = 0
    g1[5] = 704;                    // tensor_dim0_stride lo32 (= row stride)
    g1[6] = 0;                      // stride hi, tensor_dim1_stride lo = 0
    g1[7] = 0;
    v4i g2 = {0, 0, 0, 0}, g3 = {0, 0, 0, 0};  // 2D tensor: groups 2/3 unused
#if __clang_major__ >= 23
    v8i g4 = {0, 0, 0, 0, 0, 0, 0, 0};
    __builtin_amdgcn_tensor_load_to_lds(g0, g1, g2, g3, g4, 0);
#else
    __builtin_amdgcn_tensor_load_to_lds(g0, g1, g2, g3, 0);
#endif
    __builtin_amdgcn_s_wait_tensorcnt(0);
  }
  __syncthreads();

  v8f acc = {};
  const _Float16* pa = p.X + (size_t)(m0 + li) * 96 + hf * 16;
  const _Float16* pw = &lw[li * 704 + hf * 16];  // LDS-resident weight row
  for (int k = 0; k < 3; ++k) {                  // x part (K = 96)
    acc = wmma_f16(ldfragA(pa), *(const v16h*)pw, acc);
    pa += 32;
    pw += 32;
  }
  pa = p.H + (size_t)(m0 + li) * 608 + hf * 16;
  for (int k = 0; k < 19; ++k) {                 // h part (K = 608)
    acc = wmma_f16(ldfragA(pa), *(const v16h*)pw, acc);
    pa += 32;
    pw += 32;
  }
  float b = p.bc[col];
#pragma unroll
  for (int v = 0; v < 8; ++v) {
    int r = m0 + v + 8 * hf;
    float val = acc[v] + b;
    p.Hout[(size_t)r * 608 + col] = (_Float16)fmaxf(val, 0.f);
  }
}

// ---------------- phase 2: per-timestep stage kernels ----------------

struct StepP {
  // f16 padded weights (row n = output col n, stride = padded K)
  const _Float16 *cWz, *tWg1, *tWh1, *tWmu, *cWmu, *cWsig, *tWg2, *tWh2, *tWsig,
      *eW1, *eW2, *eW3;
  // f32 padded biases
  const float *cbz, *tbg1, *tbh1, *tbmu, *cbmu, *cbsig, *tbg2, *tbh2, *tbsig,
      *eb1, *eb2, *eb3;
  const _Float16* HS;  // (T,B,608) backward RNN states, f16
  const float *mb, *noise, *mask;
  float* out;  // (B,) elbo accumulator == d_out
  // activation buffers
  _Float16 *hc, *g1, *h1p, *hr, *h1, *h2, *z16;
  float *zmu, *pmu, *z32;
  int t;
};

// S1: from z_prev -> hc (tanh + 0.5*(..+h_t)), g1=relu, h1p=relu, zmu=z@Wmu^T+b
// 71 N-tiles x 16 M-tiles = 1136 tiles, K=128 (4 k-tiles)
__global__ void k_s1(StepP p) {
  int lane = threadIdx.x & 31;
  int tile = blockIdx.x * 8 + (threadIdx.x >> 5);
  int li = lane & 15, hf = lane >> 4;
  int m0 = (tile & 15) * 16, ni = tile >> 4;
  int arow = m0 + li;
  if (ni < 38) {  // hc segment, N=608
    int col = ni * 16 + li;
    v8f acc = {};
    acc = gemm_acc(p.z16, 128, p.cWz, 128, arow, col, 4, acc, hf);
    float b = p.cbz[col];
    const _Float16* hst = p.HS + (size_t)p.t * BB * 608;
#pragma unroll
    for (int v = 0; v < 8; ++v) {
      int r = m0 + v + 8 * hf;
      float hcv = 0.5f * (tanhf(acc[v] + b) + (float)hst[(size_t)r * 608 + col]);
      p.hc[(size_t)r * 608 + col] = (_Float16)hcv;
    }
  } else if (ni < 51) {  // g1 = relu(z@tWg1^T + b), N=208 (200 valid)
    int col = (ni - 38) * 16 + li;
    v8f acc = {};
    acc = gemm_acc(p.z16, 128, p.tWg1, 128, arow, col, 4, acc, hf);
    float b = p.tbg1[col];
#pragma unroll
    for (int v = 0; v < 8; ++v) {
      int r = m0 + v + 8 * hf;
      p.g1[(size_t)r * 224 + col] = (_Float16)fmaxf(acc[v] + b, 0.f);
    }
  } else if (ni < 64) {  // h1p = relu(z@tWh1^T + b)
    int col = (ni - 51) * 16 + li;
    v8f acc = {};
    acc = gemm_acc(p.z16, 128, p.tWh1, 128, arow, col, 4, acc, hf);
    float b = p.tbh1[col];
#pragma unroll
    for (int v = 0; v < 8; ++v) {
      int r = m0 + v + 8 * hf;
      p.h1p[(size_t)r * 224 + col] = (_Float16)fmaxf(acc[v] + b, 0.f);
    }
  } else {  // zmu = z_prev @ t_Wmu^T + t_bmu (kept faithful), N=112 (100 valid)
    int col = (ni - 64) * 16 + li;
    v8f acc = {};
    acc = gemm_acc(p.z16, 128, p.tWmu, 128, arow, col, 4, acc, hf);
    float b = p.tbmu[col];
    bool valid = (col < ZZ);
#pragma unroll
    for (int v = 0; v < 8; ++v) {
      int r = m0 + v + 8 * hf;
      p.zmu[(size_t)r * 128 + col] = valid ? (acc[v] + b) : 0.f;
    }
  }
}

// S2: segA: q_mu/q_sig from hc (K=608) -> z_t, -lognormal_q (via eps identity)
//     segB: g/h_ from g1,h1p (K=224) -> p_mu, hr=relu(h_)
// 14 N-tiles x 16 = 224 tiles
__global__ void k_s2(StepP p) {
  int lane = threadIdx.x & 31;
  int tile = blockIdx.x * 8 + (threadIdx.x >> 5);
  int li = lane & 15, hf = lane >> 4;
  int m0 = (tile & 15) * 16, ni = tile >> 4;
  int arow = m0 + li;
  if (ni < 7) {
    int col = ni * 16 + li;
    v8f amu = {}, asig = {};
    amu = gemm_acc(p.hc, 608, p.cWmu, 608, arow, col, 19, amu, hf);
    asig = gemm_acc(p.hc, 608, p.cWsig, 608, arow, col, 19, asig, hf);
    float bmu = p.cbmu[col], bsig = p.cbsig[col];
    bool valid = (col < ZZ);
#pragma unroll
    for (int v = 0; v < 8; ++v) {
      int r = m0 + v + 8 * hf;
      float qmu = amu[v] + bmu;
      float qsig = softplusf(asig[v] + bsig);
      float eps =
          valid ? p.noise[(size_t)r * TT * ZZ + (size_t)p.t * ZZ + col] : 0.f;
      float z = qmu + qsig * eps;
      p.z16[(size_t)r * 128 + col] = (_Float16)(valid ? z : 0.f);
      p.z32[(size_t)r * 128 + col] = valid ? z : 0.f;
      // lognormal(z, q_mu, q_sig): (z-mu)/sig == eps
      float tq = valid ? (-0.5f * eps * eps - logf(qsig) - HALF_LOG_2PI) : 0.f;
      float s = rowsum16(tq);
      if (li == 0)
        atomicAdd(&p.out[r], -p.mask[(size_t)r * TT + p.t] * s);  // minus lnq
    }
  } else {
    int col = (ni - 7) * 16 + li;
    v8f ag = {}, ah = {};
    ag = gemm_acc(p.g1, 224, p.tWg2, 224, arow, col, 7, ag, hf);
    ah = gemm_acc(p.h1p, 224, p.tWh2, 224, arow, col, 7, ah, hf);
    float bg = p.tbg2[col], bh = p.tbh2[col];
    bool valid = (col < ZZ);
#pragma unroll
    for (int v = 0; v < 8; ++v) {
      int r = m0 + v + 8 * hf;
      float g = sigmoidf(ag[v] + bg);
      float hh = ah[v] + bh;
      float pm = (1.f - g) * p.zmu[(size_t)r * 128 + col] + g * hh;
      p.pmu[(size_t)r * 128 + col] = valid ? pm : 0.f;
      p.hr[(size_t)r * 128 + col] = (_Float16)(valid ? fmaxf(hh, 0.f) : 0.f);
    }
  }
}

// S3: h1 = relu(z@eW1^T+b);  p_sig = softplus(hr@tWsig^T+b) + lognormal_p
// 14 N-tiles x 16 = 224 tiles, K=128
__global__ void k_s3(StepP p) {
  int lane = threadIdx.x & 31;
  int tile = blockIdx.x * 8 + (threadIdx.x >> 5);
  int li = lane & 15, hf = lane >> 4;
  int m0 = (tile & 15) * 16, ni = tile >> 4;
  int arow = m0 + li;
  if (ni < 7) {
    int col = ni * 16 + li;
    v8f acc = {};
    acc = gemm_acc(p.z16, 128, p.eW1, 128, arow, col, 4, acc, hf);
    float b = p.eb1[col];
    bool valid = (col < ZZ);
#pragma unroll
    for (int v = 0; v < 8; ++v) {
      int r = m0 + v + 8 * hf;
      float val = fmaxf(acc[v] + b, 0.f);
      p.h1[(size_t)r * 128 + col] = (_Float16)(valid ? val : 0.f);
    }
  } else {
    int col = (ni - 7) * 16 + li;
    v8f acc = {};
    acc = gemm_acc(p.hr, 128, p.tWsig, 128, arow, col, 4, acc, hf);
    float b = p.tbsig[col];
    bool valid = (col < ZZ);
#pragma unroll
    for (int v = 0; v < 8; ++v) {
      int r = m0 + v + 8 * hf;
      float psig = softplusf(acc[v] + b);
      float term = 0.f;
      if (valid) {
        float z = p.z32[(size_t)r * 128 + col];
        float pm = p.pmu[(size_t)r * 128 + col];
        float d = (z - pm) / psig;
        term = -0.5f * d * d - logf(psig) - HALF_LOG_2PI;
      }
      float s = rowsum16(term);
      if (li == 0) atomicAdd(&p.out[r], p.mask[(size_t)r * TT + p.t] * s);
    }
  }
}

// S4: h2 = relu(h1@eW2^T + b)  (7 N-tiles x 16 = 112 tiles)
__global__ void k_s4(StepP p) {
  int lane = threadIdx.x & 31;
  int tile = blockIdx.x * 8 + (threadIdx.x >> 5);
  int li = lane & 15, hf = lane >> 4;
  int m0 = (tile & 15) * 16, ni = tile >> 4;
  int arow = m0 + li;
  int col = ni * 16 + li;
  v8f acc = {};
  acc = gemm_acc(p.h1, 128, p.eW2, 128, arow, col, 4, acc, hf);
  float b = p.eb2[col];
  bool valid = (col < ZZ);
#pragma unroll
  for (int v = 0; v < 8; ++v) {
    int r = m0 + v + 8 * hf;
    float val = fmaxf(acc[v] + b, 0.f);
    p.h2[(size_t)r * 128 + col] = (_Float16)(valid ? val : 0.f);
  }
}

// S5: ps = clip(sigmoid(h2@eW3^T + b)); Bernoulli log-likelihood -> out
// 6 N-tiles x 16 = 96 tiles
__global__ void k_s5(StepP p) {
  int lane = threadIdx.x & 31;
  int tile = blockIdx.x * 8 + (threadIdx.x >> 5);
  int li = lane & 15, hf = lane >> 4;
  int m0 = (tile & 15) * 16, ni = tile >> 4;
  int arow = m0 + li;
  int col = ni * 16 + li;
  v8f acc = {};
  acc = gemm_acc(p.h2, 128, p.eW3, 128, arow, col, 4, acc, hf);
  float b = p.eb3[col];
  bool valid = (col < DD);
#pragma unroll
  for (int v = 0; v < 8; ++v) {
    int r = m0 + v + 8 * hf;
    float ps = sigmoidf(acc[v] + b);
    ps = fminf(fmaxf(ps, 1e-6f), 1.f - 1e-6f);
    float term = 0.f;
    if (valid) {
      float x = p.mb[(size_t)r * TT * DD + (size_t)p.t * DD + col];
      term = x * logf(ps) + (1.f - x) * log1pf(-ps);
    }
    float s = rowsum16(term);
    if (li == 0) atomicAdd(&p.out[r], p.mask[(size_t)r * TT + p.t] * s);
  }
}

// ---------------- host driver ----------------

extern "C" void kernel_launch(void* const* d_in, const int* in_sizes, int n_in,
                              void* d_out, int out_size, void* d_ws, size_t ws_size,
                              hipStream_t stream) {
  (void)in_sizes; (void)n_in; (void)out_size; (void)ws_size;
  const float* mb      = (const float*)d_in[0];
  const float* noise   = (const float*)d_in[1];
  const float* mask    = (const float*)d_in[2];
  const float* rnn_Wih = (const float*)d_in[3];
  const float* rnn_bih = (const float*)d_in[4];
  const float* rnn_Whh = (const float*)d_in[5];
  const float* rnn_bhh = (const float*)d_in[6];
  const float* h0      = (const float*)d_in[7];
  const float* c_Wz    = (const float*)d_in[8];
  const float* c_bz    = (const float*)d_in[9];
  const float* c_Wmu   = (const float*)d_in[10];
  const float* c_bmu   = (const float*)d_in[11];
  const float* c_Wsig  = (const float*)d_in[12];
  const float* c_bsig  = (const float*)d_in[13];
  const float* e_W1    = (const float*)d_in[14];
  const float* e_b1    = (const float*)d_in[15];
  const float* e_W2    = (const float*)d_in[16];
  const float* e_b2    = (const float*)d_in[17];
  const float* e_W3    = (const float*)d_in[18];
  const float* e_b3    = (const float*)d_in[19];
  const float* t_Wg1   = (const float*)d_in[20];
  const float* t_bg1   = (const float*)d_in[21];
  const float* t_Wg2   = (const float*)d_in[22];
  const float* t_bg2   = (const float*)d_in[23];
  const float* t_Wh1   = (const float*)d_in[24];
  const float* t_bh1   = (const float*)d_in[25];
  const float* t_Wh2   = (const float*)d_in[26];
  const float* t_bh2   = (const float*)d_in[27];
  const float* t_Wmu   = (const float*)d_in[28];
  const float* t_bmu   = (const float*)d_in[29];
  const float* t_Wsig  = (const float*)d_in[30];
  const float* t_bsig  = (const float*)d_in[31];
  const float* z_q_0   = (const float*)d_in[32];

  // ---- carve workspace ----
  char* base = (char*)d_ws;
  size_t off = 0;
  auto alloc = [&](size_t bytes) -> void* {
    off = (off + 255) & ~(size_t)255;
    void* p = base + off;
    off += bytes;
    return p;
  };
  _Float16* X16  = (_Float16*)alloc((size_t)TT * BB * 96 * 2);
  _Float16* HS   = (_Float16*)alloc((size_t)TT * BB * 608 * 2);
  _Float16* H016 = (_Float16*)alloc((size_t)BB * 608 * 2);
  _Float16* Wc   = (_Float16*)alloc((size_t)608 * 704 * 2);
  float*    bc   = (float*)alloc(608 * 4);
  _Float16* cWz16   = (_Float16*)alloc((size_t)608 * 128 * 2);
  _Float16* tWg1_16 = (_Float16*)alloc((size_t)224 * 128 * 2);
  _Float16* tWh1_16 = (_Float16*)alloc((size_t)224 * 128 * 2);
  _Float16* tWmu16  = (_Float16*)alloc((size_t)112 * 128 * 2);
  _Float16* cWmu16  = (_Float16*)alloc((size_t)112 * 608 * 2);
  _Float16* cWsig16 = (_Float16*)alloc((size_t)112 * 608 * 2);
  _Float16* tWg2_16 = (_Float16*)alloc((size_t)112 * 224 * 2);
  _Float16* tWh2_16 = (_Float16*)alloc((size_t)112 * 224 * 2);
  _Float16* tWsig16 = (_Float16*)alloc((size_t)112 * 128 * 2);
  _Float16* eW1_16  = (_Float16*)alloc((size_t)112 * 128 * 2);
  _Float16* eW2_16  = (_Float16*)alloc((size_t)112 * 128 * 2);
  _Float16* eW3_16  = (_Float16*)alloc((size_t)96 * 128 * 2);
  float* cbz_p  = (float*)alloc(608 * 4);
  float* tbg1_p = (float*)alloc(224 * 4);
  float* tbh1_p = (float*)alloc(224 * 4);
  float* tbmu_p = (float*)alloc(112 * 4);
  float* cbmu_p = (float*)alloc(112 * 4);
  float* cbsig_p= (float*)alloc(112 * 4);
  float* tbg2_p = (float*)alloc(112 * 4);
  float* tbh2_p = (float*)alloc(112 * 4);
  float* tbsig_p= (float*)alloc(112 * 4);
  float* eb1_p  = (float*)alloc(112 * 4);
  float* eb2_p  = (float*)alloc(112 * 4);
  float* eb3_p  = (float*)alloc(96 * 4);
  _Float16* hc16  = (_Float16*)alloc((size_t)BB * 608 * 2);
  _Float16* g1_16 = (_Float16*)alloc((size_t)BB * 224 * 2);
  _Float16* h1p16 = (_Float16*)alloc((size_t)BB * 224 * 2);
  _Float16* hr16  = (_Float16*)alloc((size_t)BB * 128 * 2);
  _Float16* h1_16 = (_Float16*)alloc((size_t)BB * 128 * 2);
  _Float16* h2_16 = (_Float16*)alloc((size_t)BB * 128 * 2);
  _Float16* z16   = (_Float16*)alloc((size_t)BB * 128 * 2);
  float* zmu32 = (float*)alloc((size_t)BB * 128 * 4);
  float* pmu32 = (float*)alloc((size_t)BB * 128 * 4);
  float* z32   = (float*)alloc((size_t)BB * 128 * 4);

  // ---- setup: zero, convert, pad ----
  hipMemsetAsync(d_out, 0, 256 * 4, stream);
  hipMemsetAsync(hc16, 0, (size_t)BB * 608 * 2, stream);
  hipMemsetAsync(g1_16, 0, (size_t)BB * 224 * 2, stream);
  hipMemsetAsync(h1p16, 0, (size_t)BB * 224 * 2, stream);
  hipMemsetAsync(hr16, 0, (size_t)BB * 128 * 2, stream);
  hipMemsetAsync(h1_16, 0, (size_t)BB * 128 * 2, stream);
  hipMemsetAsync(h2_16, 0, (size_t)BB * 128 * 2, stream);
  hipMemsetAsync(z16, 0, (size_t)BB * 128 * 2, stream);
  hipMemsetAsync(zmu32, 0, (size_t)BB * 128 * 4, stream);
  hipMemsetAsync(pmu32, 0, (size_t)BB * 128 * 4, stream);
  hipMemsetAsync(z32, 0, (size_t)BB * 128 * 4, stream);

  auto cvt = [&](const float* s, _Float16* d, int r, int c, int rp, int cw, int ds) {
    int n = rp * cw;
    k_cvt_pad<<<(n + 255) / 256, 256, 0, stream>>>(s, d, r, c, rp, cw, ds);
  };
  auto padb = [&](const float* s, float* d, int n, int np) {
    k_pad_bias<<<(np + 255) / 256, 256, 0, stream>>>(s, d, n, np);
  };

  {
    size_t n = (size_t)TT * BB * 96;
    k_cvt_x<<<(int)((n + 255) / 256), 256, 0, stream>>>(mb, X16);
  }
  k_bcast_h0<<<(BB * 608 + 255) / 256, 256, 0, stream>>>(h0, H016);
  k_bcast_z0<<<(BB * 128 + 255) / 256, 256, 0, stream>>>(z_q_0, z16, z32);

  cvt(rnn_Wih, Wc,       600,  88, 608,  96, 704);
  cvt(rnn_Whh, Wc + 96,  600, 600, 608, 608, 704);
  k_bias2<<<(608 + 255) / 256, 256, 0, stream>>>(rnn_bih, rnn_bhh, bc, 600, 608);
  cvt(c_Wz,   cWz16,   600, 100, 608, 128, 128);
  cvt(t_Wg1,  tWg1_16, 200, 100, 224, 128, 128);
  cvt(t_Wh1,  tWh1_16, 200, 100, 224, 128, 128);
  cvt(t_Wmu,  tWmu16,  100, 100, 112, 128, 128);
  cvt(c_Wmu,  cWmu16,  100, 600, 112, 608, 608);
  cvt(c_Wsig, cWsig16, 100, 600, 112, 608, 608);
  cvt(t_Wg2,  tWg2_16, 100, 200, 112, 224, 224);
  cvt(t_Wh2,  tWh2_16, 100, 200, 112, 224, 224);
  cvt(t_Wsig, tWsig16, 100, 100, 112, 128, 128);
  cvt(e_W1,   eW1_16,  100, 100, 112, 128, 128);
  cvt(e_W2,   eW2_16,  100, 100, 112, 128, 128);
  cvt(e_W3,   eW3_16,   88, 100,  96, 128, 128);
  padb(c_bz, cbz_p, 600, 608);
  padb(t_bg1, tbg1_p, 200, 224);
  padb(t_bh1, tbh1_p, 200, 224);
  padb(t_bmu, tbmu_p, 100, 112);
  padb(c_bmu, cbmu_p, 100, 112);
  padb(c_bsig, cbsig_p, 100, 112);
  padb(t_bg2, tbg2_p, 100, 112);
  padb(t_bh2, tbh2_p, 100, 112);
  padb(t_bsig, tbsig_p, 100, 112);
  padb(e_b1, eb1_p, 100, 112);
  padb(e_b2, eb2_p, 100, 112);
  padb(e_b3, eb3_p, 88, 96);

  // ---- phase 1: backward RNN scan (t = T-1 .. 0) ----
  for (int k = 0; k < TT; ++k) {
    int t = TT - 1 - k;
    RnnP rp;
    rp.X = X16 + (size_t)t * BB * 96;
    rp.H = (k == 0) ? H016 : (HS + (size_t)(t + 1) * BB * 608);
    rp.Wc = Wc;
    rp.bc = bc;
    rp.Hout = HS + (size_t)t * BB * 608;
    k_rnn<<<76, 256, 0, stream>>>(rp);   // 608 tiles, 8 waves/WG
  }

  // ---- phase 2: forward latent scan ----
  StepP sp;
  sp.cWz = cWz16; sp.tWg1 = tWg1_16; sp.tWh1 = tWh1_16; sp.tWmu = tWmu16;
  sp.cWmu = cWmu16; sp.cWsig = cWsig16; sp.tWg2 = tWg2_16; sp.tWh2 = tWh2_16;
  sp.tWsig = tWsig16; sp.eW1 = eW1_16; sp.eW2 = eW2_16; sp.eW3 = eW3_16;
  sp.cbz = cbz_p; sp.tbg1 = tbg1_p; sp.tbh1 = tbh1_p; sp.tbmu = tbmu_p;
  sp.cbmu = cbmu_p; sp.cbsig = cbsig_p; sp.tbg2 = tbg2_p; sp.tbh2 = tbh2_p;
  sp.tbsig = tbsig_p; sp.eb1 = eb1_p; sp.eb2 = eb2_p; sp.eb3 = eb3_p;
  sp.HS = HS; sp.mb = mb; sp.noise = noise; sp.mask = mask;
  sp.out = (float*)d_out;
  sp.hc = hc16; sp.g1 = g1_16; sp.h1p = h1p16; sp.hr = hr16;
  sp.h1 = h1_16; sp.h2 = h2_16; sp.z16 = z16;
  sp.zmu = zmu32; sp.pmu = pmu32; sp.z32 = z32;

  for (int t = 0; t < TT; ++t) {
    sp.t = t;
    k_s1<<<142, 256, 0, stream>>>(sp);  // 1136 tiles
    k_s2<<<28, 256, 0, stream>>>(sp);   // 224 tiles
    k_s3<<<28, 256, 0, stream>>>(sp);   // 224 tiles
    k_s4<<<14, 256, 0, stream>>>(sp);   // 112 tiles
    k_s5<<<12, 256, 0, stream>>>(sp);   // 96 tiles
  }
}
